// Channel_14302241096093
// MI455X (gfx1250) — matrix operand order repaired and checked by reading, not compiled, not run on hardware
//
#include <hip/hip_runtime.h>
#include <hip/hip_bf16.h>
#include <math.h>

// ---------------------------------------------------------------------------
// Complex FIR (full conv, 32 taps) via block-Toeplitz f16 WMMA with hi/lo
// fp32-split operands (near-fp32 accuracy), + 64-pt DFT of taps.
// Shapes: input (256,64,2048,2) f32, cof (256,64,32,2) f32, M=64.
// Out: conv (256,64,2079,2) then H_t (256,64,64,2), concatenated flat.
// ---------------------------------------------------------------------------

typedef __attribute__((ext_vector_type(16))) _Float16     v16h;
typedef __attribute__((ext_vector_type(8)))  float        v8f;
typedef __attribute__((ext_vector_type(4)))  float        f32x4;
typedef __attribute__((ext_vector_type(4)))  unsigned int u32x4;

#define SEQS   16384              // N*P
#define SMK    2048               // signal length (complex)
#define LTAPS  32                 // taps (complex)
#define OUTC   (SMK + LTAPS - 1)  // 2079 complex outputs / sequence
#define OUTR   (2 * OUTC)         // 4158 floats / sequence
#define MFFT   64
#define PADF   32                 // front halo (complex)
#define PADT   2224               // 32 + 2048 + 144 back halo (complex)
#define NTILES 17                 // ceil(2079 / 128)
#define WPB    2                  // waves (=sequences) per block

union HalfPack { _Float16 h[2]; unsigned int u; };
union BPack    { u32x4 q[2]; v16h v; };

__device__ __forceinline__ void split_pack(float re, float im,
                                           unsigned int &hi, unsigned int &lo) {
    HalfPack H, L;
    _Float16 hr = (_Float16)re;
    _Float16 hm = (_Float16)im;
    H.h[0] = hr;  H.h[1] = hm;
    L.h[0] = (_Float16)(re - (float)hr);
    L.h[1] = (_Float16)(im - (float)hm);
    hi = H.u;  lo = L.u;
}

#define WMMA_ACC(Areg, Breg, Creg)                                            \
    Creg = __builtin_amdgcn_wmma_f32_16x16x32_f16(                            \
        false, Areg, false, Breg, (short)0, Creg, false, false)

__global__ __launch_bounds__(32 * WPB)
void conv_wmma_kernel(const float* __restrict__ x,
                      const float* __restrict__ cof,
                      float* __restrict__ out)
{
    // Per-wave staging: signal split into f16 hi/lo, packed (re,im) per u32.
    __shared__ unsigned int sh_hi[WPB][PADT];
    __shared__ unsigned int sh_lo[WPB][PADT];
    __shared__ float        sh_taps[WPB][2 * LTAPS];

    const int lane = threadIdx.x & 31;
    const int wave = threadIdx.x >> 5;
    const int seq  = blockIdx.x * WPB + wave;

    unsigned int* hiBuf = sh_hi[wave];
    unsigned int* loBuf = sh_lo[wave];
    float*        taps  = sh_taps[wave];

    // ---- stage signal (fp32 -> packed f16 hi/lo) with zero halos ----------
    const float* xs = x + (size_t)seq * (2 * SMK);
    for (int i = lane; i < PADF; i += 32)            { hiBuf[i] = 0u; loBuf[i] = 0u; }
    for (int i = PADF + SMK + lane; i < PADT; i += 32){ hiBuf[i] = 0u; loBuf[i] = 0u; }
    for (int i = lane; i < SMK / 2; i += 32) {       // 2 complex per float4
        f32x4 v = reinterpret_cast<const f32x4*>(xs)[i];
        unsigned int h0, l0, h1, l1;
        split_pack(v.x, v.y, h0, l0);
        split_pack(v.z, v.w, h1, l1);
        hiBuf[PADF + 2 * i]     = h0;  loBuf[PADF + 2 * i]     = l0;
        hiBuf[PADF + 2 * i + 1] = h1;  loBuf[PADF + 2 * i + 1] = l1;
    }
    // ---- stage taps --------------------------------------------------------
    const float* cs = cof + (size_t)seq * (2 * LTAPS);
    taps[lane]      = cs[lane];
    taps[lane + 32] = cs[lane + 32];
    __syncthreads();

    // ---- build block-Toeplitz A chunks (hi/lo), fixed for whole sequence ---
    // Row i = 2*p + c  (p = complex output offset 0..7, c = re/im part).
    // A[i][t] pairs with B[t][j] = x2[2*(m0+8j+16(q-2)) + t]  (t parity = re/im).
    const int i_row = lane & 15;
    const int p     = i_row >> 1;
    const int cpart = i_row & 1;
    const int kbase = (lane >> 4) << 4;       // A K-layout: lanes>=16 start at K=16

    v16h Ah[3], Al[3];
#pragma unroll
    for (int q = 0; q < 3; ++q) {
#pragma unroll
        for (int e = 0; e < 16; ++e) {
            const int t      = kbase + e;  // K index of elem e
            const int sigrel = 16 * (q - 2) + (t >> 1);
            const int k      = p - sigrel;                      // tap index
            const int d      = t & 1;                           // signal re/im
            float val = 0.0f;
            if (k >= 0 && k < LTAPS) {
                const float hr = taps[2 * k], hm = taps[2 * k + 1];
                val = (cpart == 0) ? ((d == 0) ? hr : -hm)
                                   : ((d == 0) ? hm :  hr);
            }
            const _Float16 vh = (_Float16)val;
            Ah[q][e] = vh;
            Al[q][e] = (_Float16)(val - (float)vh);
        }
    }

    // ---- tile loop: 128 complex outputs per tile ---------------------------
    const int ncol = lane & 15;   // B/D column
    const int kh   = lane >> 4;   // B K-half / D row-half
    float* os = out + (size_t)seq * OUTR;

    for (int tile = 0; tile < NTILES; ++tile) {
        const int m0 = tile * 128;
        v8f acc = {};
#pragma unroll
        for (int q = 0; q < 3; ++q) {
            // 8 consecutive packed complex values == this lane's 16 B halves,
            // in the same contiguous k order as the A chunks (hi & lo variants).
            const int base = m0 + 8 * ncol + 16 * q + 8 * kh;   // padded complex idx
            BPack Bh, Bl;
            Bh.q[0] = *reinterpret_cast<const u32x4*>(&hiBuf[base]);
            Bh.q[1] = *reinterpret_cast<const u32x4*>(&hiBuf[base + 4]);
            Bl.q[0] = *reinterpret_cast<const u32x4*>(&loBuf[base]);
            Bl.q[1] = *reinterpret_cast<const u32x4*>(&loBuf[base + 4]);
            WMMA_ACC(Ah[q], Bh.v, acc);   // hi*hi
            WMMA_ACC(Ah[q], Bl.v, acc);   // hi*lo
            WMMA_ACC(Al[q], Bh.v, acc);   // lo*hi  (lo*lo negligible: ~2^-22)
        }
        // D: lane holds 8 consecutive floats of column ncol (row-half kh).
        const int Rbase = 2 * m0 + 16 * ncol + 8 * kh;
        if (Rbase + 8 <= OUTR) {
            f32x4 lo4 = { acc[0], acc[1], acc[2], acc[3] };
            f32x4 hi4 = { acc[4], acc[5], acc[6], acc[7] };
            *reinterpret_cast<f32x4*>(&os[Rbase])     = lo4;
            *reinterpret_cast<f32x4*>(&os[Rbase + 4]) = hi4;
        } else {
#pragma unroll
            for (int v = 0; v < 8; ++v)
                if (Rbase + v < OUTR) os[Rbase + v] = acc[v];
        }
    }
}

// ---------------------------------------------------------------------------
// H_t: 64-point DFT of zero-padded 32 complex taps, per sequence (tiny).
// ---------------------------------------------------------------------------
__global__ __launch_bounds__(256)
void dft_kernel(const float* __restrict__ cof, float* __restrict__ ht)
{
    const int gid = blockIdx.x * blockDim.x + threadIdx.x;
    if (gid >= SEQS * MFFT) return;
    const int seq = gid >> 6;
    const int m   = gid & 63;
    const float* cs = cof + (size_t)seq * (2 * LTAPS);
    const float step = -6.283185307179586f / 64.0f;
    float re = 0.0f, im = 0.0f;
#pragma unroll 4
    for (int l = 0; l < LTAPS; ++l) {
        const int idx = (m * l) & 63;          // exact angle reduction
        float s, c;
        __sincosf(step * (float)idx, &s, &c);
        const float hr = cs[2 * l], hm = cs[2 * l + 1];
        re += hr * c - hm * s;                 // (hr + i*hm) * e^{-i*2pi*m*l/64}
        im += hr * s + hm * c;
    }
    ht[2 * gid]     = re;
    ht[2 * gid + 1] = im;
}

// ---------------------------------------------------------------------------
extern "C" void kernel_launch(void* const* d_in, const int* in_sizes, int n_in,
                              void* d_out, int out_size, void* d_ws, size_t ws_size,
                              hipStream_t stream)
{
    const float* x   = (const float*)d_in[0];   // input  (256,64,2048,2) f32
    const float* cof = (const float*)d_in[1];   // cof    (256,64,32,2)   f32
    float* out = (float*)d_out;

    conv_wmma_kernel<<<SEQS / WPB, 32 * WPB, 0, stream>>>(x, cof, out);

    float* ht = out + (size_t)SEQS * OUTR;      // H_t region after conv output
    dft_kernel<<<(SEQS * MFFT + 255) / 256, 256, 0, stream>>>(cof, ht);
}
